// DA_HGAN_35948876268086
// MI455X (gfx1250) — 8-bit matrix operand order repaired and checked by reading, compiled, not run on hardware
//
#include <hip/hip_runtime.h>
#include <hip/hip_bf16.h>
#include <stdint.h>


typedef float v2f __attribute__((ext_vector_type(2)));
typedef float v8f __attribute__((ext_vector_type(8)));
typedef int   v8i __attribute__((ext_vector_type(8)));

#define N_NODES 8192
#define M_EDGES 4096
#define D_FEAT  256
#define SIGMA   0.5f
#define LRELU_A 0.2f

__device__ __forceinline__ float lrelu(float x) { return x > 0.f ? x : LRELU_A * x; }
__device__ __forceinline__ float eluf (float x) { return x > 0.f ? x : (expf(x) - 1.f); }

__device__ __forceinline__ unsigned fkey(float f) {
  unsigned u = __float_as_uint(f);
  return (u & 0x80000000u) ? ~u : (u | 0x80000000u);
}
__device__ __forceinline__ float funkey(unsigned u) {
  return (u & 0x80000000u) ? __uint_as_float(u & 0x7fffffffu) : __uint_as_float(~u);
}
__device__ __forceinline__ float wave_sum(float v) {
#pragma unroll
  for (int m = 16; m; m >>= 1) v += __shfl_xor(v, m, 32);
  return v;
}
__device__ __forceinline__ float wave_max(float v) {
#pragma unroll
  for (int m = 16; m; m >>= 1) v = fmaxf(v, __shfl_xor(v, m, 32));
  return v;
}

__device__ __forceinline__ void async_copy_b128(void* lds_dst, const void* gsrc) {
  uint32_t lds_off = (uint32_t)(uintptr_t)lds_dst;
  uint64_t gaddr   = (uint64_t)(uintptr_t)gsrc;
  asm volatile("global_load_async_to_lds_b128 %0, %1, off"
               :: "v"(lds_off), "v"(gaddr) : "memory");
}
__device__ __forceinline__ void wait_async0() {
  asm volatile("s_wait_asynccnt 0x0" ::: "memory");
}

__global__ void k_init(unsigned* slots) {
  if (threadIdx.x == 0) {
    slots[0] = fkey( __builtin_inff());
    slots[1] = fkey(-__builtin_inff());
    slots[2] = fkey( __builtin_inff());
    slots[3] = fkey(-__builtin_inff());
    slots[4] = fkey(-__builtin_inff());
    slots[5] = fkey(-__builtin_inff());
    slots[6] = fkey(-__builtin_inff());
    slots[7] = fkey(-__builtin_inff());
  }
}

__global__ void k_h8(const float* __restrict__ Hf, unsigned char* __restrict__ H8) {
  size_t tot = (size_t)N_NODES * M_EDGES;
  for (size_t i = (size_t)blockIdx.x * 256u + threadIdx.x; i < tot;
       i += (size_t)gridDim.x * 256u)
    H8[i] = Hf[i] > 0.f ? 1u : 0u;
}

__global__ __launch_bounds__(256) void k_xnorm(const float* __restrict__ X, float* __restrict__ s) {
  int row  = blockIdx.x * 8 + (threadIdx.x >> 5);
  int lane = threadIdx.x & 31;
  const float* xr = X + (size_t)row * D_FEAT;
  float acc = 0.f;
#pragma unroll
  for (int k = lane; k < D_FEAT; k += 32) { float v = xr[k]; acc += v * v; }
  acc = wave_sum(acc);
  if (lane == 0) s[row] = acc;
}

__global__ __launch_bounds__(256) void k_rho(const float* __restrict__ X,
                                             const unsigned char* __restrict__ H8,
                                             const float* __restrict__ s,
                                             float* __restrict__ rho_x) {
  __shared__ unsigned char lH[16 * M_EDGES];
  __shared__ float         lX[16 * D_FEAT];
  const int ti   = blockIdx.x;
  const int wave = threadIdx.x >> 5;
  const int lane = threadIdx.x & 31;
  const int mrow = lane & 15;
  const int half = lane >> 4;

  {
    const char* gH = (const char*)(H8 + (size_t)ti * 16 * M_EDGES);
    for (int c = threadIdx.x; c < (16 * M_EDGES) / 16; c += 256)
      async_copy_b128(&lH[c * 16], gH + (size_t)c * 16);
    const char* gX = (const char*)(X + (size_t)ti * 16 * D_FEAT);
    for (int c = threadIdx.x; c < (16 * D_FEAT * 4) / 16; c += 256)
      async_copy_b128(&lX[c * 4], gX + (size_t)c * 16);
  }
  wait_async0();
  __syncthreads();

  float sA[8];
#pragma unroll
  for (int r = 0; r < 8; ++r) sA[r] = s[ti * 16 + r + half * 8];

  float acc[8];
#pragma unroll
  for (int r = 0; r < 8; ++r) acc[r] = 0.f;

  const uint32_t* aH = (const uint32_t*)&lH[mrow * M_EDGES];
  const float*    aX = &lX[mrow * D_FEAT];

  for (int tj = wave; tj < N_NODES / 16; tj += 8) {
    const float*         bX = X  + (size_t)(tj * 16 + mrow) * D_FEAT;
    const unsigned char* bH = H8 + (size_t)(tj * 16 + mrow) * M_EDGES;
    if (tj + 8 < N_NODES / 16) {
      __builtin_prefetch(X  + (size_t)((tj + 8) * 16 + mrow) * D_FEAT, 0, 1);
      __builtin_prefetch(H8 + (size_t)((tj + 8) * 16 + mrow) * M_EDGES, 0, 1);
    }

    v8f cF = {};
#pragma unroll 4
    for (int k0 = 0; k0 < D_FEAT; k0 += 4) {
      const int ka = k0 + half * 2;
      const float2 av = *(const float2*)&aX[ka];
      const float2 bv = *(const float2*)&bX[ka];
      v2f a; a[0] = av.x; a[1] = av.y;
      v2f b; b[0] = bv.x; b[1] = bv.y;
      cF = __builtin_amdgcn_wmma_f32_16x16x4_f32(false, a, false, b, (short)0, cF, false, false);
    }

    v8i cI = {};
#pragma unroll 2
    for (int k0 = 0; k0 < M_EDGES; k0 += 64) {
      const int ka = k0 + half * 16;
      const int kb = k0 + half * 16;
      v8i a, b;
      a[0] = (int)aH[(ka +  0) >> 2];
      a[1] = (int)aH[(ka +  4) >> 2];
      a[2] = (int)aH[(ka +  8) >> 2];
      a[3] = (int)aH[(ka + 12) >> 2];
      a[4] = (int)aH[(ka + 32) >> 2];
      a[5] = (int)aH[(ka + 36) >> 2];
      a[6] = (int)aH[(ka + 40) >> 2];
      a[7] = (int)aH[(ka + 44) >> 2];
      const uint4 b03 = *(const uint4*)&bH[kb];
      const uint4 b47 = *(const uint4*)&bH[kb + 32];
      b[0] = (int)b03.x; b[1] = (int)b03.y; b[2] = (int)b03.z; b[3] = (int)b03.w;
      b[4] = (int)b47.x; b[5] = (int)b47.y; b[6] = (int)b47.z; b[7] = (int)b47.w;
      cI = __builtin_amdgcn_wmma_i32_16x16x64_iu8(false, a, false, b, cI, false, false);
    }

    const float sj = s[tj * 16 + mrow];
#pragma unroll
    for (int r = 0; r < 8; ++r) {
      const int i = ti * 16 + r + half * 8;
      const int j = tj * 16 + mrow;
      const float cosim = cF[r] * rsqrtf(sA[r] * sj);
      if (cosim > SIGMA && cI[r] > 0 && i != j) acc[r] += cosim;
    }
  }

#pragma unroll
  for (int r = 0; r < 8; ++r) {
    float v = acc[r];
    v += __shfl_xor(v, 1, 32); v += __shfl_xor(v, 2, 32);
    v += __shfl_xor(v, 4, 32); v += __shfl_xor(v, 8, 32);
    acc[r] = v;
  }
  __syncthreads();
  float* red = (float*)lH;
  if (mrow == 0) {
#pragma unroll
    for (int r = 0; r < 8; ++r) red[wave * 16 + r + half * 8] = acc[r];
  }
  __syncthreads();
  if (threadIdx.x < 16) {
    float t = 0.f;
#pragma unroll
    for (int w = 0; w < 8; ++w) t += red[w * 16 + threadIdx.x];
    rho_x[ti * 16 + threadIdx.x] = t;
  }
}

__global__ void k_rho_e(const unsigned char* __restrict__ H8, const float* __restrict__ rho_x,
                        float* __restrict__ rho_e) {
  int k = blockIdx.x * 256 + threadIdx.x;
  float acc = 0.f;
  for (int i = 0; i < N_NODES; ++i)
    if (H8[(size_t)i * M_EDGES + k]) acc += rho_x[i];
  rho_e[k] = acc;
}

__global__ void k_minmax(const float* rho_x, const float* rho_e, unsigned* slots) {
  int i = blockIdx.x * 256 + threadIdx.x;
  if (i < N_NODES) {
    unsigned key = fkey(rho_x[i]);
    atomicMin(&slots[0], key);
    atomicMax(&slots[1], key);
  }
  if (i < M_EDGES) {
    unsigned key = fkey(rho_e[i]);
    atomicMin(&slots[2], key);
    atomicMax(&slots[3], key);
  }
}

__global__ __launch_bounds__(256) void k_xw_ew(const float* __restrict__ X,
                                               const float* __restrict__ E,
                                               const float* __restrict__ W,
                                               float* __restrict__ XW, float* __restrict__ EW) {
  __shared__ float lW[D_FEAT * 8];
  for (int c = threadIdx.x; c < (D_FEAT * 8 * 4) / 16; c += 256)
    async_copy_b128(&lW[c * 4], (const char*)W + (size_t)c * 16);
  wait_async0();
  __syncthreads();

  int g    = blockIdx.x * 8 + (threadIdx.x >> 5);
  int lane = threadIdx.x & 31;
  int mrow = lane & 15, half = lane >> 4;
  const float* src; float* dst; int rowbase;
  if (g < N_NODES / 16) { src = X; dst = XW; rowbase = g * 16; }
  else                  { src = E; dst = EW; rowbase = (g - N_NODES / 16) * 16; }

  v8f c = {};
#pragma unroll 4
  for (int k0 = 0; k0 < D_FEAT; k0 += 4) {
    const int ka = k0 + half * 2;
    const float2 av = *(const float2*)&src[(size_t)(rowbase + mrow) * D_FEAT + ka];
    v2f a; a[0] = av.x; a[1] = av.y;
    v2f b;
    b[0] = (mrow < 8) ? lW[ka * 8 + mrow]       : 0.f;
    b[1] = (mrow < 8) ? lW[(ka + 1) * 8 + mrow] : 0.f;
    c = __builtin_amdgcn_wmma_f32_16x16x4_f32(false, a, false, b, (short)0, c, false, false);
  }
  if (mrow < 8) {
#pragma unroll
    for (int r = 0; r < 8; ++r)
      dst[(size_t)(rowbase + r + half * 8) * 8 + mrow] = c[r];
  }
}

__global__ void k_uv(const float* XW, const float* EW, const float* ax, const float* ae,
                     float* u1, float* v2, float* v1, float* u2, unsigned* slots) {
  int i = blockIdx.x * 256 + threadIdx.x;
  if (i < N_NODES) {
    float a = 0.f, b = 0.f;
#pragma unroll
    for (int c = 0; c < 8; ++c) { float w = XW[i * 8 + c]; a += w * ax[c]; b += w * ae[8 + c]; }
    u1[i] = a; v2[i] = b;
    atomicMax(&slots[4], fkey(a));
    atomicMax(&slots[7], fkey(b));
  }
  if (i < M_EDGES) {
    float a = 0.f, b = 0.f;
#pragma unroll
    for (int c = 0; c < 8; ++c) { float w = EW[i * 8 + c]; a += w * ax[8 + c]; b += w * ae[c]; }
    v1[i] = a; u2[i] = b;
    atomicMax(&slots[5], fkey(a));
    atomicMax(&slots[6], fkey(b));
  }
}

__global__ void k_rt(const float* rho_x, const float* rho_e, const unsigned* slots,
                     float* rt1, float* rt2) {
  int i = blockIdx.x * 256 + threadIdx.x;
  if (i < N_NODES) {
    float rmin = funkey(slots[0]), rmax = funkey(slots[1]);
    float amax = lrelu(funkey(slots[4]) + funkey(slots[5]));
    rt1[i] = (rho_x[i] - rmin) / (rmax - rmin) * amax;
  }
  if (i < M_EDGES) {
    float rmin = funkey(slots[2]), rmax = funkey(slots[3]);
    float amax = lrelu(funkey(slots[6]) + funkey(slots[7]));
    rt2[i] = (rho_e[i] - rmin) / (rmax - rmin) * amax;
  }
}

__global__ __launch_bounds__(256) void k_stats1(const unsigned char* __restrict__ H8,
                                                const float* u1, const float* v1,
                                                const float* rt1, float* M1, float* Z1) {
  int i    = blockIdx.x * 8 + (threadIdx.x >> 5);
  int lane = threadIdx.x & 31;
  const unsigned char* hr = H8 + (size_t)i * M_EDGES;
  float ui = u1[i], rti = rt1[i];
  float m = -__builtin_inff();
  for (int k = lane; k < M_EDGES; k += 32)
    if (hr[k]) m = fmaxf(m, lrelu(ui + v1[k]));
  m = wave_max(m);
  float z = 0.f;
  for (int k = lane; k < M_EDGES; k += 32)
    if (hr[k]) z += expf(lrelu(ui + v1[k]) - m);
  z = wave_sum(z);
  if (lane == 0) { M1[i] = m + rti; Z1[i] = z; }
}

__global__ void k_stats2(const unsigned char* __restrict__ H8, const float* u2,
                         const float* v2, const float* rt2, float* M2, float* Z2) {
  int k = blockIdx.x * 256 + threadIdx.x;
  float uk = u2[k], rtk = rt2[k];
  float m = -__builtin_inff();
  for (int i = 0; i < N_NODES; ++i)
    if (H8[(size_t)i * M_EDGES + k]) m = fmaxf(m, lrelu(uk + v2[i]));
  float z = 0.f;
  for (int i = 0; i < N_NODES; ++i)
    if (H8[(size_t)i * M_EDGES + k]) z += expf(lrelu(uk + v2[i]) - m);
  M2[k] = m + rtk; Z2[k] = z;
}

__global__ void k_agg1(const unsigned char* __restrict__ H8, const float* u1, const float* v1,
                       const float* rt1, const float* M1, const float* Z1,
                       const float* XW, float* Enew) {
  int k = blockIdx.x * 256 + threadIdx.x;
  float vk = v1[k];
  float acc[8];
#pragma unroll
  for (int c = 0; c < 8; ++c) acc[c] = 0.f;
  for (int i = 0; i < N_NODES; ++i) {
    if (H8[(size_t)i * M_EDGES + k]) {
      float w = expf(lrelu(u1[i] + vk) + rt1[i] - M1[i]) / Z1[i];
#pragma unroll
      for (int c = 0; c < 8; ++c) acc[c] += w * XW[i * 8 + c];
    }
  }
#pragma unroll
  for (int c = 0; c < 8; ++c) Enew[k * 8 + c] = eluf(acc[c]);
}

__global__ __launch_bounds__(256) void k_agg2(const unsigned char* __restrict__ H8,
                                              const float* u2, const float* v2,
                                              const float* rt2, const float* M2,
                                              const float* Z2, const float* Enew,
                                              float* __restrict__ out) {
  int i    = blockIdx.x * 8 + (threadIdx.x >> 5);
  int lane = threadIdx.x & 31;
  const unsigned char* hr = H8 + (size_t)i * M_EDGES;
  float vi = v2[i];
  float acc[8];
#pragma unroll
  for (int c = 0; c < 8; ++c) acc[c] = 0.f;
  for (int k = lane; k < M_EDGES; k += 32) {
    if (hr[k]) {
      float w = expf(lrelu(u2[k] + vi) + rt2[k] - M2[k]) / Z2[k];
#pragma unroll
      for (int c = 0; c < 8; ++c) acc[c] += w * Enew[k * 8 + c];
    }
  }
#pragma unroll
  for (int c = 0; c < 8; ++c) acc[c] = wave_sum(acc[c]);
  if (lane == 0) {
#pragma unroll
    for (int c = 0; c < 8; ++c) out[(size_t)i * 8 + c] = eluf(acc[c]);
  }
}

extern "C" void kernel_launch(void* const* d_in, const int* in_sizes, int n_in,
                              void* d_out, int out_size, void* d_ws, size_t ws_size,
                              hipStream_t stream) {
  (void)in_sizes; (void)n_in; (void)out_size; (void)ws_size;
  const float* X  = (const float*)d_in[0];
  const float* E  = (const float*)d_in[1];
  const float* Hf = (const float*)d_in[2];
  const float* W  = (const float*)d_in[3];
  const float* ax = (const float*)d_in[4];
  const float* ae = (const float*)d_in[5];
  float* out = (float*)d_out;

  char*  base = (char*)d_ws;
  size_t off  = 0;
  auto alloc = [&](size_t bytes) -> void* {
    void* r = base + off;
    off = (off + bytes + 255) & ~(size_t)255;
    return r;
  };
  unsigned char* H8   = (unsigned char*)alloc((size_t)N_NODES * M_EDGES);
  float* s     = (float*)alloc(N_NODES * 4);
  float* rho_x = (float*)alloc(N_NODES * 4);
  float* rho_e = (float*)alloc(M_EDGES * 4);
  float* XW    = (float*)alloc((size_t)N_NODES * 8 * 4);
  float* EW    = (float*)alloc((size_t)M_EDGES * 8 * 4);
  float* u1    = (float*)alloc(N_NODES * 4);
  float* v1    = (float*)alloc(M_EDGES * 4);
  float* u2    = (float*)alloc(M_EDGES * 4);
  float* v2    = (float*)alloc(N_NODES * 4);
  float* rt1   = (float*)alloc(N_NODES * 4);
  float* rt2   = (float*)alloc(M_EDGES * 4);
  float* M1    = (float*)alloc(N_NODES * 4);
  float* Z1    = (float*)alloc(N_NODES * 4);
  float* M2    = (float*)alloc(M_EDGES * 4);
  float* Z2    = (float*)alloc(M_EDGES * 4);
  float* Enew  = (float*)alloc((size_t)M_EDGES * 8 * 4);
  unsigned* slots = (unsigned*)alloc(64);

  k_init   <<<1,    32,  0, stream>>>(slots);
  k_h8     <<<4096, 256, 0, stream>>>(Hf, H8);
  k_xnorm  <<<1024, 256, 0, stream>>>(X, s);
  k_rho    <<<512,  256, 0, stream>>>(X, H8, s, rho_x);
  k_rho_e  <<<16,   256, 0, stream>>>(H8, rho_x, rho_e);
  k_minmax <<<32,   256, 0, stream>>>(rho_x, rho_e, slots);
  k_xw_ew  <<<96,   256, 0, stream>>>(X, E, W, XW, EW);
  k_uv     <<<32,   256, 0, stream>>>(XW, EW, ax, ae, u1, v2, v1, u2, slots);
  k_rt     <<<32,   256, 0, stream>>>(rho_x, rho_e, slots, rt1, rt2);
  k_stats1 <<<1024, 256, 0, stream>>>(H8, u1, v1, rt1, M1, Z1);
  k_stats2 <<<16,   256, 0, stream>>>(H8, u2, v2, rt2, M2, Z2);
  k_agg1   <<<16,   256, 0, stream>>>(H8, u1, v1, rt1, M1, Z1, XW, Enew);
  k_agg2   <<<1024, 256, 0, stream>>>(H8, u2, v2, rt2, M2, Z2, Enew, out);
}
